// EncoderLayer_53618371723384
// MI455X (gfx1250) — compile-verified, matrix operand order checked by reading, not run on hardware
//
#include <hip/hip_runtime.h>
#include <hip/hip_bf16.h>
#include <stdint.h>

typedef __bf16 bf16_t;
typedef __attribute__((ext_vector_type(16))) __bf16 v16bf;
typedef __attribute__((ext_vector_type(8)))  float  v8f;
typedef __attribute__((ext_vector_type(4)))  int    v4i;

#define D_MODEL 1024
#define N_HEAD  16
#define D_HEAD  64
#define D_FF    4096
#define SEQ     2048
#define BATCH   4
#define NROW    (SEQ * BATCH)   /* 8192 token rows */
#define LN_EPS  1e-5f

// ---------------------------------------------------------------------------
// CDNA5 async global->LDS path (ASYNCcnt), with portable fallback.
// Builtin signature (from hipcc diagnostic): (as1 int4*, as3 int4*, Ii, Ii).
// ---------------------------------------------------------------------------
#if defined(__has_builtin)
#if __has_builtin(__builtin_amdgcn_global_load_async_to_lds_b128) && \
    __has_builtin(__builtin_amdgcn_s_wait_asynccnt)
#define CDNA5_ASYNC 1
#endif
#endif

#if defined(CDNA5_ASYNC)
typedef __attribute__((address_space(1))) v4i as1_v4i;
typedef __attribute__((address_space(3))) v4i as3_v4i;
#endif

__device__ inline void copy16_g2l(const bf16_t* g, bf16_t* l) {
#if defined(CDNA5_ASYNC)
  __builtin_amdgcn_global_load_async_to_lds_b128(
      (as1_v4i*)(bf16_t*)g, (as3_v4i*)l, /*offset=*/0, /*cpol=*/0);
#else
  *(uint4*)l = *(const uint4*)g;
#endif
}

// Wait for own async copies, then make all waves' copies visible.
__device__ inline void async_join() {
#if defined(CDNA5_ASYNC)
  __builtin_amdgcn_s_wait_asynccnt(0);
#endif
  __syncthreads();
}

// ---------------------------------------------------------------------------
// helpers
// ---------------------------------------------------------------------------
__device__ inline v8f v8f_zero() {
  v8f r;
#pragma unroll
  for (int i = 0; i < 8; ++i) r[i] = 0.0f;
  return r;
}

// CDNA5 WMMA bf16: D(f32 16x16) = A(16x32) * B(32x16) + C
__device__ inline v8f wmma_bf16(v16bf a, v16bf b, v8f c) {
  return __builtin_amdgcn_wmma_f32_16x16x32_bf16(
      /*neg_a=*/false, a, /*neg_b=*/false, b,
      /*c_mod=*/(short)0, c, /*reuse_a=*/false, /*reuse_b=*/false);
}

// A-fragment (16x32, 16-bit) per ISA 7.12.2: lane holds row M=(lane&15);
// pairs (a[2i],a[2i+1]) = row[ks], ks = (i<4?0:16) + khalf + 2*(i&3),
// khalf = (lane>=16)*8  =>  two contiguous 16B chunks: [khalf..+7],[khalf+16..+23].
// B-fragment for "X @ W^T" is identical with row = N index (W row-major).
__device__ inline v16bf load_row_frag(const bf16_t* rowbase, int khalf) {
  union { uint4 u[2]; v16bf v; } r;
  r.u[0] = *(const uint4*)(rowbase + khalf);
  r.u[1] = *(const uint4*)(rowbase + khalf + 16);
  return r.v;
}

// ---------------------------------------------------------------------------
// f32 -> bf16 conversion (RNE)
// ---------------------------------------------------------------------------
__global__ void f32_to_bf16_kernel(const float* __restrict__ in,
                                   bf16_t* __restrict__ out, int n) {
  int i = blockIdx.x * blockDim.x + threadIdx.x;
  int stride = gridDim.x * blockDim.x;
  for (; i < n; i += stride) out[i] = (bf16_t)in[i];
}

// ---------------------------------------------------------------------------
// Tiled GEMM: C[M,N] = A[M,K] @ W[N,K]^T  (+bias)(+residual)(+relu)
// Block 128x256, 256 threads = 8 waves (2x4), wave tile 64x64 (16 accums),
// double-buffered LDS with async global->LDS copies.
// M mult of 128, N mult of 256, K mult of 32 (true for all uses here).
// ---------------------------------------------------------------------------
template <bool BIAS, bool RELU, bool RESID, bool OUTBF16>
__global__ __launch_bounds__(256) void gemm_wt_kernel(
    const bf16_t* __restrict__ A, const bf16_t* __restrict__ W,
    const float* __restrict__ bias, const float* __restrict__ resid,
    void* __restrict__ Cout, int M, int N, int K) {
  constexpr int LDT = 40;  // 32-wide K tile padded; 80B pitch keeps 16B align
  __shared__ bf16_t As[2][128 * LDT];
  __shared__ bf16_t Ws[2][256 * LDT];

  const int tid   = threadIdx.x;
  const int lane  = tid & 31;
  const int wv    = tid >> 5;          // 0..7
  const int wm    = wv >> 2;           // 0..1
  const int wn    = wv & 3;            // 0..3
  const int m0    = blockIdx.y * 128;
  const int n0    = blockIdx.x * 256;
  const int khalf = (lane >> 4) << 3;  // 0 or 8
  const int lr    = lane & 15;
  const int rowh  = (lane >> 4) * 8;

  v8f acc[4][4];
#pragma unroll
  for (int i = 0; i < 4; ++i)
#pragma unroll
    for (int j = 0; j < 4; ++j) acc[i][j] = v8f_zero();

  const int NK = K >> 5;

  auto issue_tile = [&](int kt, int buf) {
    const int k0 = kt << 5;
#pragma unroll
    for (int p = 0; p < 2; ++p) {          // A: 128x32 = 512 x 16B
      int a = tid + (p << 8);
      int r = a >> 2, sg = (a & 3) << 3;
      copy16_g2l(&A[(size_t)(m0 + r) * K + k0 + sg], &As[buf][r * LDT + sg]);
    }
#pragma unroll
    for (int p = 0; p < 4; ++p) {          // W: 256x32 = 1024 x 16B
      int a = tid + (p << 8);
      int r = a >> 2, sg = (a & 3) << 3;
      copy16_g2l(&W[(size_t)(n0 + r) * K + k0 + sg], &Ws[buf][r * LDT + sg]);
    }
  };

  int cur = 0;
  issue_tile(0, 0);
  async_join();

  for (int kt = 0; kt < NK; ++kt) {
    if (kt + 1 < NK) issue_tile(kt + 1, cur ^ 1);  // prefetch next K tile

    v16bf af[4], wf[4];
#pragma unroll
    for (int i = 0; i < 4; ++i) {
      af[i] = load_row_frag(&As[cur][(wm * 64 + i * 16 + lr) * LDT], khalf);
      wf[i] = load_row_frag(&Ws[cur][(wn * 64 + i * 16 + lr) * LDT], khalf);
    }
#pragma unroll
    for (int mi = 0; mi < 4; ++mi)
#pragma unroll
      for (int ni = 0; ni < 4; ++ni)
        acc[mi][ni] = wmma_bf16(af[mi], wf[ni], acc[mi][ni]);

    async_join();  // next tile landed; also fences buffer reuse
    cur ^= 1;
  }

  // Epilogue. C layout: lane holds col n=(lane&15), rows rowh+0..7 per acc reg.
#pragma unroll
  for (int mi = 0; mi < 4; ++mi) {
#pragma unroll
    for (int ni = 0; ni < 4; ++ni) {
      int col = n0 + wn * 64 + ni * 16 + lr;
      float bv = BIAS ? bias[col] : 0.0f;
#pragma unroll
      for (int v = 0; v < 8; ++v) {
        int row = m0 + wm * 64 + mi * 16 + rowh + v;
        float x = acc[mi][ni][v] + bv;
        if (RESID) x += resid[(size_t)row * N + col];
        if (RELU) x = fmaxf(x, 0.0f);
        if (OUTBF16)
          ((bf16_t*)Cout)[(size_t)row * N + col] = (bf16_t)x;
        else
          ((float*)Cout)[(size_t)row * N + col] = x;
      }
    }
  }
}

// ---------------------------------------------------------------------------
// Flash attention: one WG (128 thr = 4 waves) per (head-batch, 64-query tile).
// Wave owns 16 query rows. Scores never touch HBM (online softmax).
// V is stored TRANSPOSED in LDS so P@V B-fragments are two ds_load_b128.
// qkv layout: (L, B, 3*D) bf16; head n=(b,h) slices e = [h*64, h*64+64).
// ---------------------------------------------------------------------------
__global__ __launch_bounds__(128) void flash_attn_kernel(
    const bf16_t* __restrict__ qkv, const uint8_t* __restrict__ mask,
    bf16_t* __restrict__ outp) {
  constexpr int LDQ  = 72;   // 64-wide dh tile padded (144B pitch)
  constexpr int LDVT = 136;  // 128-wide k tile padded (272B pitch)
  constexpr int LDP  = 136;
  __shared__ bf16_t Qs[64 * LDQ];        //  9.0 KB
  __shared__ bf16_t Ks[128 * LDQ];       // 18.0 KB
  __shared__ bf16_t Vt[64 * LDVT];       // 17.0 KB  (V^T: [d][k])
  __shared__ bf16_t Ps[4][16 * LDP];     // 17.0 KB  per-wave P (16x128)

  const int tid   = threadIdx.x;
  const int lane  = tid & 31;
  const int wv    = tid >> 5;          // 0..3
  const int lr    = lane & 15;
  const int khalf = (lane >> 4) << 3;  // 0 or 8
  const int rowh  = (lane >> 4) * 8;

  const int n  = blockIdx.y;           // 0..63 head-batches
  const int b  = n >> 4;
  const int h  = n & 15;
  const int q0 = blockIdx.x * 64;

  const size_t SB = (size_t)BATCH * 3 * D_MODEL;  // 12288 stride over l
  const bf16_t* qb = qkv + (size_t)b * 3 * D_MODEL + h * D_HEAD;
  const bf16_t* kb = qb + D_MODEL;
  const bf16_t* vb = qb + 2 * D_MODEL;

  // Q tile (64x64): 512 x 16B over 128 threads (async; joined below)
#pragma unroll
  for (int p = 0; p < 4; ++p) {
    int idx = tid + p * 128;
    int r = idx >> 3;
    int c = (idx & 7) << 3;
    copy16_g2l(&qb[(size_t)(q0 + r) * SB + c], &Qs[r * LDQ + c]);
  }

  float rowm[8], rowl[8];
  v8f oacc[4];
#pragma unroll
  for (int v = 0; v < 8; ++v) { rowm[v] = -__builtin_inff(); rowl[v] = 0.0f; }
#pragma unroll
  for (int t = 0; t < 4; ++t) oacc[t] = v8f_zero();

  for (int kt = 0; kt < SEQ / 128; ++kt) {
    __syncthreads();  // previous iteration readers of Ks/Vt done
#pragma unroll
    for (int p = 0; p < 8; ++p) {
      int idx = tid + p * 128;
      int r = idx >> 3;           // key row in tile
      int c = (idx & 7) << 3;     // d chunk
      size_t g = (size_t)(kt * 128 + r) * SB + c;
      copy16_g2l(&kb[g], &Ks[r * LDQ + c]);       // K row-major (async)
      union { uint4 u; bf16_t e[8]; } vvv;        // V transposed (sync)
      vvv.u = *(const uint4*)(&vb[g]);
#pragma unroll
      for (int j = 0; j < 8; ++j) Vt[(c + j) * LDVT + r] = vvv.e[j];
    }
    async_join();  // Q (first iter) + K async landed; Vt fenced by barrier

    // S = Q K^T : 16x128 per wave (8 N-tiles x 2 K-steps)
    v8f sacc[8];
#pragma unroll
    for (int t = 0; t < 8; ++t) sacc[t] = v8f_zero();
#pragma unroll
    for (int ks = 0; ks < 64; ks += 32) {
      v16bf aq = load_row_frag(&Qs[(wv * 16 + lr) * LDQ + ks], khalf);
#pragma unroll
      for (int t = 0; t < 8; ++t) {
        v16bf bk = load_row_frag(&Ks[(t * 16 + lr) * LDQ + ks], khalf);
        sacc[t] = wmma_bf16(aq, bk, sacc[t]);
      }
    }

    // scale + key padding mask
#pragma unroll
    for (int t = 0; t < 8; ++t) {
      int kk = kt * 128 + t * 16 + lr;
      bool mk = mask[b * SEQ + kk] != 0;
#pragma unroll
      for (int v = 0; v < 8; ++v) {
        float s = sacc[t][v] * 0.125f;  // 1/sqrt(64)
        sacc[t][v] = mk ? -__builtin_inff() : s;
      }
    }

    // online softmax per row (row lives in one 16-lane half -> shfl width 16)
#pragma unroll
    for (int v = 0; v < 8; ++v) {
      float tm = sacc[0][v];
#pragma unroll
      for (int t = 1; t < 8; ++t) tm = fmaxf(tm, sacc[t][v]);
#pragma unroll
      for (int off = 8; off >= 1; off >>= 1)
        tm = fmaxf(tm, __shfl_xor(tm, off, 16));
      float newm  = fmaxf(rowm[v], tm);
      bool  inff  = (newm == -__builtin_inff());
      float alpha = inff ? 1.0f : __expf(rowm[v] - newm);
      float ts = 0.0f;
#pragma unroll
      for (int t = 0; t < 8; ++t) {
        float pe = inff ? 0.0f : __expf(sacc[t][v] - newm);
        ts += pe;
        Ps[wv][(rowh + v) * LDP + t * 16 + lr] = (bf16_t)pe;
      }
#pragma unroll
      for (int off = 8; off >= 1; off >>= 1) ts += __shfl_xor(ts, off, 16);
      rowl[v] = rowl[v] * alpha + ts;
      rowm[v] = newm;
#pragma unroll
      for (int t = 0; t < 4; ++t) oacc[t][v] *= alpha;
    }
    __syncthreads();  // P visible (C-layout -> A-layout via LDS)

    // O += P @ V : (16x128)*(128x64); V^T row = output col -> vector frags
#pragma unroll
    for (int ks0 = 0; ks0 < 128; ks0 += 32) {
      v16bf ap = load_row_frag(&Ps[wv][lr * LDP + ks0], khalf);
#pragma unroll
      for (int t = 0; t < 4; ++t) {
        v16bf bv = load_row_frag(&Vt[(t * 16 + lr) * LDVT + ks0], khalf);
        oacc[t] = wmma_bf16(ap, bv, oacc[t]);
      }
    }
  }

  // normalize + write (L,B,D) bf16
  bf16_t* ob = outp + (size_t)b * D_MODEL + h * D_HEAD;
#pragma unroll
  for (int v = 0; v < 8; ++v) {
    float inv = 1.0f / rowl[v];
    int l = q0 + wv * 16 + rowh + v;
#pragma unroll
    for (int t = 0; t < 4; ++t) {
      int d = t * 16 + lr;
      ob[(size_t)l * (BATCH * D_MODEL) + d] = (bf16_t)(oacc[t][v] * inv);
    }
  }
}

// ---------------------------------------------------------------------------
// LayerNorm over D_MODEL, one row per 256-thread block; optional bf16 copy.
// ---------------------------------------------------------------------------
template <bool EMIT_BF16>
__global__ __launch_bounds__(256) void layernorm_kernel(
    const float* __restrict__ x, const float* __restrict__ g,
    const float* __restrict__ be, float* __restrict__ y,
    bf16_t* __restrict__ yb) {
  __shared__ float red[256];
  const int row = blockIdx.x;
  const int tid = threadIdx.x;
  const float* xr = x + (size_t)row * D_MODEL;

  float xs[4];
  float s = 0.0f;
#pragma unroll
  for (int i = 0; i < 4; ++i) { xs[i] = xr[tid + i * 256]; s += xs[i]; }
  red[tid] = s;
  __syncthreads();
  for (int o = 128; o > 0; o >>= 1) {
    if (tid < o) red[tid] += red[tid + o];
    __syncthreads();
  }
  float mu = red[0] * (1.0f / D_MODEL);
  __syncthreads();

  float vs = 0.0f;
#pragma unroll
  for (int i = 0; i < 4; ++i) { float d = xs[i] - mu; vs += d * d; }
  red[tid] = vs;
  __syncthreads();
  for (int o = 128; o > 0; o >>= 1) {
    if (tid < o) red[tid] += red[tid + o];
    __syncthreads();
  }
  float rs = rsqrtf(red[0] * (1.0f / D_MODEL) + LN_EPS);

#pragma unroll
  for (int i = 0; i < 4; ++i) {
    int c = tid + i * 256;
    float o = (xs[i] - mu) * rs * g[c] + be[c];
    y[(size_t)row * D_MODEL + c] = o;
    if (EMIT_BF16) yb[(size_t)row * D_MODEL + c] = (bf16_t)o;
  }
}

// ---------------------------------------------------------------------------
// Orchestration
// ---------------------------------------------------------------------------
extern "C" void kernel_launch(void* const* d_in, const int* in_sizes, int n_in,
                              void* d_out, int out_size, void* d_ws,
                              size_t ws_size, hipStream_t stream) {
  const float*   src   = (const float*)d_in[0];
  const uint8_t* mask  = (const uint8_t*)d_in[1];
  const float*   w_qkv = (const float*)d_in[2];
  const float*   fc_w  = (const float*)d_in[3];
  const float*   fc_b  = (const float*)d_in[4];
  const float*   w1    = (const float*)d_in[5];
  const float*   b1    = (const float*)d_in[6];
  const float*   w2    = (const float*)d_in[7];
  const float*   b2    = (const float*)d_in[8];
  const float*   g1    = (const float*)d_in[9];
  const float*   be1   = (const float*)d_in[10];
  const float*   g2    = (const float*)d_in[11];
  const float*   be2   = (const float*)d_in[12];

  char* ws = (char*)d_ws;
  size_t off = 0;
  auto alloc = [&](size_t bytes) -> void* {
    void* p = ws + off;
    off += (bytes + 255) & ~(size_t)255;
    return p;
  };

  const size_t NE = (size_t)NROW * D_MODEL;  // 8M activations
  bf16_t* srcb  = (bf16_t*)alloc(NE * 2);
  bf16_t* wqkvb = (bf16_t*)alloc((size_t)3 * D_MODEL * D_MODEL * 2);
  bf16_t* fcwb  = (bf16_t*)alloc((size_t)D_MODEL * D_MODEL * 2);
  bf16_t* w1b   = (bf16_t*)alloc((size_t)D_FF * D_MODEL * 2);
  bf16_t* w2b   = (bf16_t*)alloc((size_t)D_MODEL * D_FF * 2);
  bf16_t* qkvb  = (bf16_t*)alloc((size_t)NROW * 3 * D_MODEL * 2);
  bf16_t* attnb = (bf16_t*)alloc(NE * 2);
  float*  x1    = (float*)alloc(NE * 4);
  float*  ln1   = (float*)alloc(NE * 4);
  bf16_t* ln1b  = (bf16_t*)alloc(NE * 2);
  bf16_t* hb    = (bf16_t*)alloc((size_t)NROW * D_FF * 2);
  float*  x2    = x1;  // x1 dead after LN1 -> reuse

  // 1) f32 -> bf16 for GEMM operands
  f32_to_bf16_kernel<<<2048, 256, 0, stream>>>(src, srcb, (int)NE);
  f32_to_bf16_kernel<<<2048, 256, 0, stream>>>(w_qkv, wqkvb, 3 * D_MODEL * D_MODEL);
  f32_to_bf16_kernel<<<1024, 256, 0, stream>>>(fc_w, fcwb, D_MODEL * D_MODEL);
  f32_to_bf16_kernel<<<2048, 256, 0, stream>>>(w1, w1b, D_FF * D_MODEL);
  f32_to_bf16_kernel<<<2048, 256, 0, stream>>>(w2, w2b, D_MODEL * D_FF);

  // 2) qkv = src @ w_qkv^T            (8192 x 3072 x 1024)
  {
    dim3 grid(3 * D_MODEL / 256, NROW / 128);
    gemm_wt_kernel<false, false, false, true><<<grid, 256, 0, stream>>>(
        srcb, wqkvb, nullptr, nullptr, qkvb, NROW, 3 * D_MODEL, D_MODEL);
  }
  // 3) flash attention -> attnb (L,B,D)
  {
    dim3 grid(SEQ / 64, BATCH * N_HEAD);
    flash_attn_kernel<<<grid, 128, 0, stream>>>(qkvb, mask, attnb);
  }
  // 4) x1 = attn @ fc_w^T + fc_b + src   (residual fused, f32 out)
  {
    dim3 grid(D_MODEL / 256, NROW / 128);
    gemm_wt_kernel<true, false, true, false><<<grid, 256, 0, stream>>>(
        attnb, fcwb, fc_b, src, x1, NROW, D_MODEL, D_MODEL);
  }
  // 5) ln1 = LayerNorm(x1) -> f32 + bf16
  layernorm_kernel<true><<<NROW, 256, 0, stream>>>(x1, g1, be1, ln1, ln1b);
  // 6) h = relu(ln1 @ w1^T + b1)       (8192 x 4096 x 1024)
  {
    dim3 grid(D_FF / 256, NROW / 128);
    gemm_wt_kernel<true, true, false, true><<<grid, 256, 0, stream>>>(
        ln1b, w1b, b1, nullptr, hb, NROW, D_FF, D_MODEL);
  }
  // 7) x2 = h @ w2^T + b2 + ln1        (8192 x 1024 x 4096)
  {
    dim3 grid(D_MODEL / 256, NROW / 128);
    gemm_wt_kernel<true, false, true, false><<<grid, 256, 0, stream>>>(
        hb, w2b, b2, ln1, x2, NROW, D_MODEL, D_FF);
  }
  // 8) out = LayerNorm(x2) -> d_out (f32)
  layernorm_kernel<false><<<NROW, 256, 0, stream>>>(x2, g2, be2, (float*)d_out,
                                                    nullptr);

  (void)in_sizes; (void)n_in; (void)out_size; (void)ws_size;
}
